// Attention_84112639525530
// MI455X (gfx1250) — compile-verified, matrix operand order checked by reading, not run on hardware
//
#include <hip/hip_runtime.h>
#include <math.h>

#define HID     1024
#define EDIM    2048      // 2*HID, the K dimension
#define BATCH   32
#define SRC_LEN 2048
#define M_TOTAL (BATCH * SRC_LEN)   // 65536 rows (m = s*32 + b)
#define KT      (EDIM / 32)         // 64 k-tiles of 32
#define NT      (HID / 16)          // 64 n-tiles of 16
#define MTILE   64                  // rows per block (4 WMMA m-tiles)
#define NSPLIT  8                   // N split into 8 slices of 128 columns

typedef __attribute__((ext_vector_type(16))) __bf16 v16bf;
typedef __attribute__((ext_vector_type(8)))  float  v8f;
typedef __attribute__((ext_vector_type(4)))  unsigned int v4u;
typedef __attribute__((ext_vector_type(8)))  int v8i;
typedef __attribute__((ext_vector_type(4)))  int v4i;

#if __has_builtin(__builtin_amdgcn_tensor_load_to_lds) && \
    __has_builtin(__builtin_amdgcn_s_wait_tensorcnt)
#define USE_TDM 1
#else
#define USE_TDM 0
#endif

// f32 -> bf16, round to nearest even
__device__ __forceinline__ unsigned short f2bf(float f) {
    unsigned int u = __float_as_uint(f);
    u += 0x7FFFu + ((u >> 16) & 1u);
    return (unsigned short)(u >> 16);
}

// gfx1250 async global->LDS copy (ASYNCcnt-tracked)
__device__ __forceinline__ void async_copy_b128(unsigned int ldsOff, unsigned long long gaddr) {
    asm volatile("global_load_async_to_lds_b128 %0, %1, off"
                 :: "v"(ldsOff), "v"(gaddr) : "memory");
}
__device__ __forceinline__ void wait_async0() {
    asm volatile("s_wait_asynccnt 0" ::: "memory");
}
__device__ __forceinline__ unsigned int lds_off(const void* p) {
    // generic pointers to LDS carry the LDS byte offset in the low 32 bits
    return (unsigned int)(unsigned long long)(uintptr_t)p;
}

#if USE_TDM
// TDM: load a 2D tile (rows x rowBytes, stride strideBytes) into LDS.
// D# per ISA 8.3/8.4: group0 = {count=1 | lds_addr | global_addr | type=2},
// group1 = {data_size=4B, tensor_dim0/1, tile_dim0/1, dim0_stride}, rest = 0.
// This toolchain exposes the 6-arg builtin: (v4u, v8i, v4i, v4i, v8i, i32 cpol).
__device__ __forceinline__ void tdm_load_2d(unsigned long long gaddr, unsigned int ldsAddr,
                                            unsigned int rowUnits, unsigned int rows,
                                            unsigned int strideUnits) {
    v4u g0 = { 1u,                                   // count=1 (valid), user mode
               ldsAddr,                              // lds_addr[31:0]
               (unsigned int)gaddr,                  // global_addr[31:0]
               (unsigned int)((gaddr >> 32) & 0x01FFFFFFull) | (2u << 30) }; // type=2
    v8i g1 = { (int)(2u << 16),                      // data_size = 2 (4 bytes)
               (int)((rowUnits & 0xFFFFu) << 16),    // tensor_dim0[15:0]
               (int)(((rowUnits >> 16) & 0xFFFFu) | ((rows & 0xFFFFu) << 16)), // dim0 hi | dim1 lo
               (int)((rowUnits & 0xFFFFu) << 16),    // dim1 hi(0) | tile_dim0
               (int)(rows & 0xFFFFu),                // tile_dim1 | tile_dim2(0)
               (int)strideUnits,                     // tensor_dim0_stride[31:0]
               0, 0 };                               // stride hi, dim1_stride
    v4i gz4 = { 0, 0, 0, 0 };
    v8i gz8 = { 0, 0, 0, 0, 0, 0, 0, 0 };
    __builtin_amdgcn_tensor_load_to_lds(g0, g1, gz4, gz4, gz8, 0);
}
#endif

// --- 16-bit WMMA operand lane layout (ISA 7.12.2) ---
// A (16x32): lane%16 <-> M, laneHalf = lane>>4 selects K sub-range.
// Element j (vgpr=j/2, p=j&1): K = (vgpr<4 ? 0 : 16) + 8*laneHalf + 2*(vgpr&3) + p
// B (32x16) mirrors with lane%16 <-> N, same per-lane K packing.

// ---------------------------------------------------------------------------
// Kernel 1: pack w_e = attn_w[:, HID:] into bf16 B-fragments, lane-linear
// (1KB per 16x32 fragment) so B tiles are contiguous per (n-tile, k-pair).
// ---------------------------------------------------------------------------
__global__ void __launch_bounds__(256) prep_we_kernel(
    const float* __restrict__ attn_w, unsigned short* __restrict__ web)
{
    int idx  = blockIdx.x * 256 + threadIdx.x;     // < NT*KT*512 = 2,097,152
    int frag = idx >> 9;
    int r    = idx & 511;
    int lane = r >> 4;
    int j    = r & 15;
    int nt   = frag >> 6;          // frag / KT
    int kt   = frag & 63;          // frag % KT
    int laneHalf = lane >> 4;
    int vg = j >> 1, p = j & 1;
    int k  = ((vg < 4) ? 0 : 16) + laneHalf * 8 + (vg & 3) * 2 + p;
    int h  = nt * 16 + (lane & 15);
    int e  = kt * 32 + k;
    web[idx] = f2bf(attn_w[(size_t)h * 3072 + HID + e]);
}

// ---------------------------------------------------------------------------
// Kernel 2: hb[b,i] = hidden[b,:] . attn_w[i,:HID] + attn_b[i]  (tiny)
// ---------------------------------------------------------------------------
__global__ void __launch_bounds__(256) hproj_kernel(
    const float* __restrict__ hidden, const float* __restrict__ attn_w,
    const float* __restrict__ attn_b, float* __restrict__ hb)
{
    int g = blockIdx.x * 256 + threadIdx.x;        // < 32768
    int b = g >> 10;
    int i = g & 1023;
    const float4* wr = (const float4*)(attn_w + (size_t)i * 3072);
    const float4* hr = (const float4*)(hidden + (size_t)b * HID);
    float s = 0.0f;
#pragma unroll 4
    for (int j = 0; j < HID / 4; ++j) {
        float4 w = wr[j], h = hr[j];
        s += w.x * h.x + w.y * h.y + w.z * h.z + w.w * h.w;
    }
    hb[g] = s + attn_b[i];
}

__global__ void __launch_bounds__(256) zero_lw_kernel(float* __restrict__ lw)
{
    lw[blockIdx.x * 256 + threadIdx.x] = 0.0f;
}

// ---------------------------------------------------------------------------
// Kernel 3: fused e_proj GEMM (bf16 WMMA) + tanh + dot(v_w) partial sums.
// Block = 64 M x 128 N x full K.  8 waves, each 1 n-tile x 4 m-tiles.
// A: enc f32 chunks double-buffered via global_load_async_to_lds_b128,
//    converted LDS->LDS into bf16 fragments.
// B: per-chunk 8x2KB tile DMA'd by wave 0 via tensor_load_to_lds (TENSORcnt),
//    double-buffered in LDS; fallback = direct global b128 loads.
// ---------------------------------------------------------------------------
__global__ void __launch_bounds__(256) attn_energy_kernel(
    const float* __restrict__ enc, const unsigned short* __restrict__ web,
    const float* __restrict__ hb, const float* __restrict__ vw,
    float* __restrict__ lw)
{
    __shared__ __align__(16) float          fbuf[2][4096];  // 2 x 16KB f32 enc chunks
    __shared__ __align__(32) unsigned short abuf[2][4096];  // 2 x 8KB bf16 A-fragments
#if USE_TDM
    __shared__ __align__(32) unsigned short bbuf[2][8192];  // 2 x 16KB bf16 B tiles
#endif
    __shared__ float red[MTILE];

    const int t        = threadIdx.x;
    const int lane     = t & 31;
    const int wv       = t >> 5;                 // wave 0..7 = n-tile within slice
    const int laneHalf = lane >> 4;
    const int laneN    = lane & 15;
    const int bid      = blockIdx.x;
    const int ng       = bid & (NSPLIT - 1);     // N slice fast -> enc L2 reuse
    const int mBase    = (bid >> 3) * MTILE;
    const int ntile    = ng * 8 + wv;            // this wave's n-tile (0..63)

    if (t < MTILE) red[t] = 0.0f;

    v8f acc[4] = {};                             // 4 m-tiles of 16x16 f32

    // async A staging: thread t moves 4 x 16B per chunk
    const int arow = t >> 4;
    const int au   = t & 15;
    const unsigned long long gbase0 = (unsigned long long)(uintptr_t)enc
        + ((unsigned long long)(mBase + arow) * EDIM + (unsigned)(au * 4)) * 4ull;

#if USE_TDM
    // B tile source for chunk kb: rows = 8 n-tiles (stride KT KB), 2KB per row
    const unsigned long long wbase = (unsigned long long)(uintptr_t)web
        + (unsigned long long)(ng * 8) * KT * 1024ull;
#endif

    // ---- prologue: chunk 0 in flight
    {
        const unsigned int l0 = lds_off(&fbuf[0][0]) + (unsigned)(t * 16);
#pragma unroll
        for (int q = 0; q < 4; ++q)
            async_copy_b128(l0 + q * 4096u, gbase0 + (unsigned long long)q * (16ull * EDIM * 4));
#if USE_TDM
        if (wv == 0)
            tdm_load_2d(wbase, lds_off(&bbuf[0][0]), 512u, 8u, 16384u);
#endif
    }

    for (int kb = 0; kb < KT / 2; ++kb) {        // 32 chunks of K=64
        const int cur = kb & 1;
        wait_async0();                           // my A slice of chunk kb arrived
#if USE_TDM
        if (wv == 0) __builtin_amdgcn_s_wait_tensorcnt(0);  // B tile kb in LDS
#endif
        __syncthreads();                         // chunk kb visible; prev compute done
        if (kb + 1 < KT / 2) {                   // overlap next chunk with convert+compute
            const unsigned long long gb = gbase0 + ((unsigned long long)(kb + 1) * 64ull) * 4ull;
            const unsigned int l0 = lds_off(&fbuf[cur ^ 1][0]) + (unsigned)(t * 16);
#pragma unroll
            for (int q = 0; q < 4; ++q)
                async_copy_b128(l0 + q * 4096u, gb + (unsigned long long)q * (16ull * EDIM * 4));
#if USE_TDM
            if (wv == 0)
                tdm_load_2d(wbase + (unsigned long long)(kb + 1) * 2048ull,
                            lds_off(&bbuf[cur ^ 1][0]), 512u, 8u, 16384u);
#endif
        }
        // ---- convert 64x64 f32 chunk -> 8 bf16 A-fragments (LDS -> LDS)
#pragma unroll
        for (int c = 0; c < 4; ++c) {
            const int row = c * 16 + (t >> 4);
            const int el0 = (t & 15) * 4;
            const float4 f4 = *(const float4*)&fbuf[cur][row * 64 + el0];
            float fv[4] = {f4.x, f4.y, f4.z, f4.w};
            const int mt = row >> 4, mrow = row & 15;
#pragma unroll
            for (int q = 0; q < 4; ++q) {
                int el = el0 + q;
                int frag = mt * 2 + (el >> 5);
                int kk = el & 31;
                int vg = ((kk >> 4) & 1) * 4 + ((kk >> 1) & 3);
                int j  = vg * 2 + (kk & 1);
                int ln = ((kk >> 3) & 1) * 16 + mrow;
                abuf[cur][frag * 512 + ln * 16 + j] = f2bf(fv[q]);
            }
        }
        __syncthreads();                         // fragments visible to all waves
        // ---- preload all operands, then 8 back-to-back WMMAs
        v16bf afr[2][4], bfr[2];
#pragma unroll
        for (int f = 0; f < 2; ++f) {
#if USE_TDM
            bfr[f] = *(const v16bf*)&bbuf[cur][wv * 1024 + f * 512 + lane * 16];
#else
            const int kt = kb * 2 + f;
            bfr[f] = *(const v16bf*)(web + ((size_t)(ntile * KT + kt)) * 512 + lane * 16);
#endif
#pragma unroll
            for (int mt = 0; mt < 4; ++mt)
                afr[f][mt] = *(const v16bf*)&abuf[cur][(mt * 2 + f) * 512 + lane * 16];
        }
#pragma unroll
        for (int f = 0; f < 2; ++f)
#pragma unroll
            for (int mt = 0; mt < 4; ++mt)
                acc[mt] = __builtin_amdgcn_wmma_f32_16x16x32_bf16(
                    false, afr[f][mt], false, bfr[f], (short)0, acc[mt], false, false);
    }

    // ---- epilogue: energy = tanh(acc + hb), partial rowsum over this h-slice
    // C layout: (vgpr r, lane) = (M = r + 8*laneHalf, N = laneN)
    const int h = ntile * 16 + laneN;
    const float vwh = vw[h];
#pragma unroll
    for (int mt = 0; mt < 4; ++mt)
#pragma unroll
        for (int r = 0; r < 8; ++r) {
            int mloc = mt * 16 + r + laneHalf * 8;
            int b    = (mBase + mloc) & 31;          // m = s*32 + b
            float v  = tanhf(acc[mt][r] + hb[b * HID + h]) * vwh;
#pragma unroll
            for (int off = 8; off >= 1; off >>= 1) v += __shfl_xor(v, off, 16);
            if (laneN == 0) atomicAdd(&red[mloc], v);
        }
    __syncthreads();
    if (t < MTILE) atomicAdd(&lw[mBase + t], red[t]);   // accumulate h-slices
}

// ---------------------------------------------------------------------------
// Kernel 4: softmax over s per batch row. lw indexed by m = s*32+b.
// ---------------------------------------------------------------------------
__global__ void __launch_bounds__(256) softmax_kernel(
    const float* __restrict__ lw, float* __restrict__ out)
{
    __shared__ float red[256];
    const int b = blockIdx.x, t = threadIdx.x;
    float v[8];
    float mx = -3.0e38f;
#pragma unroll
    for (int q = 0; q < 8; ++q) {
        v[q] = lw[(t + q * 256) * 32 + b];
        mx = fmaxf(mx, v[q]);
    }
    red[t] = mx; __syncthreads();
    for (int s = 128; s > 0; s >>= 1) {
        if (t < s) red[t] = fmaxf(red[t], red[t + s]);
        __syncthreads();
    }
    mx = red[0]; __syncthreads();
    float sum = 0.0f;
#pragma unroll
    for (int q = 0; q < 8; ++q) { v[q] = __expf(v[q] - mx); sum += v[q]; }
    red[t] = sum; __syncthreads();
    for (int s = 128; s > 0; s >>= 1) {
        if (t < s) red[t] += red[t + s];
        __syncthreads();
    }
    const float inv = 1.0f / red[0];
#pragma unroll
    for (int q = 0; q < 8; ++q) out[(size_t)b * SRC_LEN + t + q * 256] = v[q] * inv;
}

// ---------------------------------------------------------------------------
// workspace: [0,4MB) bf16 w_e fragments | [4MB,+128KB) hb | +256KB logits
// ---------------------------------------------------------------------------
extern "C" void kernel_launch(void* const* d_in, const int* in_sizes, int n_in,
                              void* d_out, int out_size, void* d_ws, size_t ws_size,
                              hipStream_t stream)
{
    const float* hidden = (const float*)d_in[0];
    const float* enc    = (const float*)d_in[1];
    const float* attn_w = (const float*)d_in[2];
    const float* attn_b = (const float*)d_in[3];
    const float* vw     = (const float*)d_in[4];
    float* out = (float*)d_out;

    unsigned short* web = (unsigned short*)d_ws;                        // 4 MB
    float* hb = (float*)((char*)d_ws + (size_t)4 * 1024 * 1024);        // 128 KB
    float* lw = hb + BATCH * HID;                                       // 256 KB

    prep_we_kernel<<<(NT * KT * 512) / 256, 256, 0, stream>>>(attn_w, web);
    hproj_kernel<<<(BATCH * HID) / 256, 256, 0, stream>>>(hidden, attn_w, attn_b, hb);
    zero_lw_kernel<<<M_TOTAL / 256, 256, 0, stream>>>(lw);
    attn_energy_kernel<<<(M_TOTAL / MTILE) * NSPLIT, 256, 0, stream>>>(enc, web, hb, vw, lw);
    softmax_kernel<<<BATCH, 256, 0, stream>>>(lw, out);
}
